// ArDCA_87591563035127
// MI455X (gfx1250) — matrix operand order repaired and checked by reading, not compiled, not run on hardware
//
#include <hip/hip_runtime.h>
#include <hip/hip_bf16.h>

typedef __attribute__((ext_vector_type(16))) __bf16 v16bf;
typedef __attribute__((ext_vector_type(8)))  float  v8f;

#define LPOS 256
#define NQ   21
#define QQ   441                     // NQ*NQ
#define QP   32
#define NSEQ 8192
#define BM   256                     // sequence rows per workgroup (32 per wave)
#define NWAVE 8
#define NGEMM (LPOS * (NSEQ / BM))   // 256 * 32 = 8192 block partials
#define RB    512                    // reduce blocks
#define TILE_B   (NQ * QP * 2)       // packed bf16 tile bytes = 1344
#define PART_BYTES 40960             // partials region at start of ws (16B aligned)

__device__ __forceinline__ unsigned short f2bf(float f) {
    unsigned int x = __float_as_uint(f);
    unsigned int r = (x + 0x7FFFu + ((x >> 16) & 1u)) >> 16;
    return (unsigned short)r;
}

__device__ __forceinline__ void wait_async0() {
    asm volatile("s_wait_asynccnt 0x0" ::: "memory");
}

// Async-copy one 16B chunk of a packed bf16 J pair (2 tiles) straight to LDS.
// 168 chunks cover the pair; LDS tile stride is 2048B (a padded to 32 rows).
__device__ __forceinline__ void stage_pair_async(
        const unsigned short* __restrict__ pk, int i, int jn,
        unsigned ldsBase, int t) {
    if (t < 168 && (t < 84 ? (jn < i) : (jn + 1 < i))) {
        unsigned voff = (unsigned)((i * LPOS + jn) * TILE_B) + (unsigned)t * 16u;
        unsigned vdst = ldsBase + (unsigned)t * 16u + (unsigned)(t / 84) * 704u;
        asm volatile("global_load_async_to_lds_b128 %0, %1, %2 offset:0"
                     :: "v"(vdst), "v"(voff), "s"(pk)
                     : "memory");
    }
}

// One j-step: shared B fragments (J tile), two A one-hot fragments (32 rows).
__device__ __forceinline__ void wmma_step(
        const unsigned short* __restrict__ tile,   // 32x32 bf16, row=a, col=k
        int s0, int s1, int kbA, int kbB, int acol,
        v8f& a00, v8f& a01, v8f& a10, v8f& a11) {
    union { v16bf v; unsigned short u[16]; } A0, A1;
    #pragma unroll
    for (int e = 0; e < 16; ++e) {
        // A k order used here (contiguous, matches B): lanes0-15: K=e; lanes16-31: K=16+e
        int k = kbB + e;
        A0.u[e] = (k == s0) ? (unsigned short)0x3F80 : (unsigned short)0;
        A1.u[e] = (k == s1) ? (unsigned short)0x3F80 : (unsigned short)0;
    }
    // B k order used here (contiguous, same as A): lanes0-15 hold K=0..15, lanes16-31 K=16..31, N=lane&15
    union { v16bf v; uint4 q[2]; } B0, B1;
    B0.q[0] = *reinterpret_cast<const uint4*>(&tile[acol * QP + kbB]);
    B0.q[1] = *reinterpret_cast<const uint4*>(&tile[acol * QP + kbB + 8]);
    B1.q[0] = *reinterpret_cast<const uint4*>(&tile[(16 + acol) * QP + kbB]);
    B1.q[1] = *reinterpret_cast<const uint4*>(&tile[(16 + acol) * QP + kbB + 8]);
    a00 = __builtin_amdgcn_wmma_f32_16x16x32_bf16(false, A0.v, false, B0.v, (short)0, a00, false, false);
    a01 = __builtin_amdgcn_wmma_f32_16x16x32_bf16(false, A0.v, false, B1.v, (short)0, a01, false, false);
    a10 = __builtin_amdgcn_wmma_f32_16x16x32_bf16(false, A1.v, false, B0.v, (short)0, a10, false, false);
    a11 = __builtin_amdgcn_wmma_f32_16x16x32_bf16(false, A1.v, false, B1.v, (short)0, a11, false, false);
}

// ---------------------------------------------------------------------------
// Pre-pack: J[i,j,:,:] (j<i) -> bf16 tiles [a(21)][k(32)], k>=21 zeroed.
// ---------------------------------------------------------------------------
__global__ __launch_bounds__(256) void ardca_pack_kernel(
        const float* __restrict__ J, unsigned short* __restrict__ pk) {
    const int ii = blockIdx.x >> 8, jj = blockIdx.x & 255;
    if (jj >= ii) return;
    const float* src = J + (size_t)blockIdx.x * QQ;
    unsigned short* dst = pk + (size_t)blockIdx.x * (NQ * QP);
    for (int e = threadIdx.x; e < NQ * QP; e += 256) {
        int a = e >> 5, k = e & 31;
        dst[e] = (k < NQ) ? f2bf(src[a * NQ + k]) : (unsigned short)0;
    }
}

// ---------------------------------------------------------------------------
// Fused kernel: per (i, m-tile of 256 rows) compute contrib via bf16 WMMA over
// j<i (2 j per barrier, quad-buffered J tiles), then fused log-softmax +
// label gather + weighted sum, deterministic block partial.
// ASYNC=true: packed-bf16 J staged with global_load_async_to_lds_b128.
// ---------------------------------------------------------------------------
template <bool ASYNC>
__global__ __launch_bounds__(256) void ardca_gemm_kernel(
        const int* __restrict__ seqs, const float* __restrict__ w,
        const float* __restrict__ h, const float* __restrict__ J,
        const unsigned short* __restrict__ pk, float* __restrict__ ws) {
    __shared__ unsigned char seqLDS[BM * LPOS];                     // 64 KB
    __shared__ __align__(16) unsigned short JLDS[2][2 * QP * QP];   // 8 KB quad buffer
    __shared__ float CLDS[NWAVE][32][QP];                           // 32 KB
    __shared__ float red[256];

    const int i    = blockIdx.x;
    const int m0   = blockIdx.y * BM;
    const int tid  = threadIdx.x;
    const int wave = tid >> 5;
    const int lane = tid & 31;

    // Stage seqs tile (BM x L) packed to u8 — int4 loads, u32 stores.
    {
        const int* sp = seqs + m0 * LPOS;
        for (int e4 = tid; e4 < (BM * LPOS) / 4; e4 += 256) {
            int4 v = reinterpret_cast<const int4*>(sp)[e4];
            unsigned int p = (unsigned)(v.x & 255) | ((unsigned)(v.y & 255) << 8)
                           | ((unsigned)(v.z & 255) << 16) | ((unsigned)(v.w & 255) << 24);
            reinterpret_cast<unsigned int*>(seqLDS)[e4] = p;
        }
    }
    // Zero all 4 J tiles (padding must be exactly 0.0 — 0*Inf/NaN poisons WMMA).
    for (int e = tid; e < 2 * 2 * QP * QP; e += 256)
        (&JLDS[0][0])[e] = 0;

    // Per-thread staging slots (fallback path): pair of contiguous 441-float blocks.
    const int r0 = tid, r1 = tid + 256, r2 = tid + 512, r3 = tid + 768;
    auto slot = [](int r) {
        int blk = (r >= QQ) ? 1 : 0;
        int rr  = r - QQ * blk;
        return blk * (QP * QP) + (rr / NQ) * QP + (rr % NQ);
    };
    const int sl0 = slot(r0), sl1 = slot(r1), sl2 = slot(r2);
    const int sl3 = (r3 < 2 * QQ) ? slot(r3) : 0;
    const int blk1 = (r1 >= QQ) ? 1 : 0;

    const float* Jbase = J + (size_t)i * LPOS * QQ;
    const unsigned ldsB0 = (unsigned)(uintptr_t)(&JLDS[0][0]);
    const unsigned ldsB1 = (unsigned)(uintptr_t)(&JLDS[1][0]);
    __syncthreads();

    // Prologue: stage pair (j=0, j=1) into buffer 0.
    if (ASYNC) {
        if (i > 0) stage_pair_async(pk, i, 0, ldsB0, tid);
        wait_async0();
    } else if (i > 0) {
        const float* Jb = Jbase;
        unsigned short* dst = JLDS[0];
        dst[sl0] = f2bf(Jb[r0]);
        if (blk1 < i) dst[sl1] = f2bf(Jb[r1]);
        if (1 < i) {
            dst[sl2] = f2bf(Jb[r2]);
            if (r3 < 2 * QQ) dst[sl3] = f2bf(Jb[r3]);
        }
    }
    __syncthreads();

    const int mrow = wave * 32 + (lane & 15);   // sub-tile 0 row; sub-tile 1 = +16
    const int kbA  = (lane < 16) ? 0 : 8;
    const int kbB  = (lane < 16) ? 0 : 16;
    const int acol = lane & 15;

    v8f acc00 = {}, acc01 = {}, acc10 = {}, acc11 = {};

    for (int jp = 0; jp < i; jp += 2) {
        const int cur = (jp >> 1) & 1, nxt = cur ^ 1;
        const int jn  = jp + 2;
        const bool more = (jn < i);

        // Issue staging for the next pair early (hidden under WMMAs).
        float t0 = 0.f, t1 = 0.f, t2 = 0.f, t3 = 0.f;
        if (ASYNC) {
            if (more) stage_pair_async(pk, i, jn, nxt ? ldsB1 : ldsB0, tid);
        } else if (more) {
            const float* Jb = Jbase + (size_t)jn * QQ;   // 882 contiguous floats
            t0 = Jb[r0];
            t1 = Jb[r1];
            t2 = Jb[r2];
            if (r3 < 2 * QQ) t3 = Jb[r3];
        }

        // Compute j = jp (and jp+1 if valid): 4 (or 8) WMMAs per wave.
        {
            const unsigned short* tile0 = JLDS[cur];
            int s0 = seqLDS[mrow * LPOS + jp];
            int s1 = seqLDS[(mrow + 16) * LPOS + jp];
            wmma_step(tile0, s0, s1, kbA, kbB, acol, acc00, acc01, acc10, acc11);
            if (jp + 1 < i) {
                const unsigned short* tile1 = JLDS[cur] + QP * QP;
                int s0b = seqLDS[mrow * LPOS + jp + 1];
                int s1b = seqLDS[(mrow + 16) * LPOS + jp + 1];
                wmma_step(tile1, s0b, s1b, kbA, kbB, acol, acc00, acc01, acc10, acc11);
            }
        }

        if (ASYNC) {
            wait_async0();                 // own async stores to LDS complete
        } else if (more) {
            unsigned short* dst = JLDS[nxt];
            dst[sl0] = f2bf(t0);
            if (jn + blk1 < i) dst[sl1] = f2bf(t1);
            if (jn + 1 < i) {
                dst[sl2] = f2bf(t2);
                if (r3 < 2 * QQ) dst[sl3] = f2bf(t3);
            }
        }
        __syncthreads();
    }

    // Spill C: lane(l&15)=column, VGPR r -> row (l<16 ? r : 8+r); sub-tile1 rows +16.
    #pragma unroll
    for (int r = 0; r < 8; ++r) {
        int mr = (lane < 16) ? r : (8 + r);
        CLDS[wave][mr][acol]           = acc00[r];
        CLDS[wave][mr][16 + acol]      = acc01[r];
        CLDS[wave][16 + mr][acol]      = acc10[r];
        CLDS[wave][16 + mr][16 + acol] = acc11[r];
    }
    __syncthreads();

    // Fused epilogue: all 32 lanes, one sequence row each.
    float partial;
    {
        const int brow = wave * 32 + lane;
        const int b    = m0 + brow;
        const int si   = seqLDS[brow * LPOS + i];
        const float* hr = h + i * NQ;
        float mx = -1e30f, lsel = 0.0f;
        float logit[NQ];
        #pragma unroll
        for (int a = 0; a < NQ; ++a) {
            float v = CLDS[wave][lane][a] + hr[a];
            logit[a] = v;
            mx = fmaxf(mx, v);
        }
        float sum = 0.0f;
        #pragma unroll
        for (int a = 0; a < NQ; ++a) {
            sum += __expf(logit[a] - mx);
            if (a == si) lsel = logit[a];
        }
        float lse = mx + __logf(sum);
        partial = w[b] * (lsel - lse);               // w[b] * ll[b,i]
    }

    // Deterministic block tree-reduction -> one partial per block.
    red[tid] = partial;
    __syncthreads();
    for (int st = 128; st > 0; st >>= 1) {
        if (tid < st) red[tid] += red[tid + st];
        __syncthreads();
    }
    if (tid == 0)
        ws[blockIdx.y * gridDim.x + blockIdx.x] = red[0];
}

// ---------------------------------------------------------------------------
// Regularizers + weight sum: deterministic block partials.
// ---------------------------------------------------------------------------
__global__ __launch_bounds__(256) void ardca_reduce_kernel(
        const float* __restrict__ w, const float* __restrict__ h,
        const float* __restrict__ J, float* __restrict__ ws) {
    __shared__ float red[256];
    const int tid = threadIdx.x;
    const int bid = blockIdx.x;

    float js = 0.0f;
    for (int blk = bid; blk < LPOS * LPOS; blk += gridDim.x) {
        int ii = blk >> 8, jj = blk & 255;
        if (jj < ii) {
            const float* p = J + (size_t)blk * QQ;
            for (int e = tid; e < QQ; e += 256) { float v = p[e]; js += v * v; }
        }
    }
    float hs = 0.0f;
    for (int e = bid * 256 + tid; e < LPOS * NQ; e += gridDim.x * 256) {
        float v = h[e]; hs += v * v;
    }
    float wsum = 0.0f;
    for (int e = bid * 256 + tid; e < NSEQ; e += gridDim.x * 256) wsum += w[e];

    red[tid] = js; __syncthreads();
    for (int st = 128; st > 0; st >>= 1) { if (tid < st) red[tid] += red[tid + st]; __syncthreads(); }
    if (tid == 0) ws[NGEMM + 0 * RB + bid] = red[0];
    __syncthreads();

    red[tid] = hs; __syncthreads();
    for (int st = 128; st > 0; st >>= 1) { if (tid < st) red[tid] += red[tid + st]; __syncthreads(); }
    if (tid == 0) ws[NGEMM + 1 * RB + bid] = red[0];
    __syncthreads();

    red[tid] = wsum; __syncthreads();
    for (int st = 128; st > 0; st >>= 1) { if (tid < st) red[tid] += red[tid + st]; __syncthreads(); }
    if (tid == 0) ws[NGEMM + 2 * RB + bid] = red[0];
}

// ---------------------------------------------------------------------------
// Final deterministic assembly of (loss, nll, reg).
// ---------------------------------------------------------------------------
__global__ __launch_bounds__(256) void ardca_final_kernel(
        const float* __restrict__ ws, float* __restrict__ out) {
    __shared__ float red[256];
    __shared__ float res[4];
    const int tid = threadIdx.x;

    float s = 0.0f;
    for (int e = tid; e < NGEMM; e += 256) s += ws[e];
    red[tid] = s; __syncthreads();
    for (int st = 128; st > 0; st >>= 1) { if (tid < st) red[tid] += red[tid + st]; __syncthreads(); }
    if (tid == 0) res[0] = red[0];
    __syncthreads();

    #pragma unroll
    for (int c = 0; c < 3; ++c) {
        float t = 0.0f;
        for (int e = tid; e < RB; e += 256) t += ws[NGEMM + c * RB + e];
        red[tid] = t; __syncthreads();
        for (int st = 128; st > 0; st >>= 1) { if (tid < st) red[tid] += red[tid + st]; __syncthreads(); }
        if (tid == 0) res[1 + c] = red[0];
        __syncthreads();
    }

    if (tid == 0) {
        float Sll  = res[0];
        float js   = res[1];
        float hs   = res[2];
        float wsum = res[3];
        float Meff = fmaxf(wsum, 1e-12f);
        float nll  = -Sll / Meff;
        float reg  = 0.5f * 1e-6f * hs + 0.5f * 1e-4f * js;
        out[0] = nll + reg;
        out[1] = nll;
        out[2] = reg;
    }
}

extern "C" void kernel_launch(void* const* d_in, const int* in_sizes, int n_in,
                              void* d_out, int out_size, void* d_ws, size_t ws_size,
                              hipStream_t stream) {
    (void)in_sizes; (void)n_in; (void)out_size;
    const int*   seqs = (const int*)d_in[0];    // (M, L) int32
    const float* w    = (const float*)d_in[1];  // (M,)
    const float* h    = (const float*)d_in[2];  // (L, Q)
    const float* J    = (const float*)d_in[3];  // (L, L, Q, Q)
    float* out = (float*)d_out;                 // 3 floats: loss, nll, reg
    float* ws  = (float*)d_ws;                  // partials at offset 0

    const size_t pkBytes = (size_t)LPOS * LPOS * NQ * QP * 2;   // ~88 MB packed J
    const bool useAsync = (ws_size >= PART_BYTES + pkBytes);
    unsigned short* pk = (unsigned short*)((char*)d_ws + PART_BYTES);

    ardca_reduce_kernel<<<RB, 256, 0, stream>>>(w, h, J, ws);

    dim3 grid(LPOS, NSEQ / BM);
    if (useAsync) {
        ardca_pack_kernel<<<LPOS * LPOS, 256, 0, stream>>>(J, pk);
        ardca_gemm_kernel<true><<<grid, 256, 0, stream>>>(seqs, w, h, J, pk, ws);
    } else {
        ardca_gemm_kernel<false><<<grid, 256, 0, stream>>>(seqs, w, h, J, nullptr, ws);
    }

    ardca_final_kernel<<<1, 256, 0, stream>>>(ws, out);
}
